// EMA_67577015435824
// MI455X (gfx1250) — matrix operand order repaired and checked by reading, not compiled, not run on hardware
//
#include <hip/hip_runtime.h>
#include <hip/hip_bf16.h>
#include <math.h>

// Problem constants (from reference): B=32, HW=4096, C=256, G=32 -> CG=8, 64x64 planes.
#define HWN   4096
#define CCH   256
#define CG    8
#define Hh    64
#define Ww    64
#define EPSV  1e-5f

typedef __attribute__((ext_vector_type(16))) __bf16 v16bf;
typedef __attribute__((ext_vector_type(8)))  float  v8f;

__device__ __forceinline__ float bf2f(unsigned short u) {
    union { unsigned int i; float f; } cv; cv.i = ((unsigned int)u) << 16; return cv.f;
}
__device__ __forceinline__ unsigned short f2bf(float f) {
    __bf16 h = (__bf16)f;                       // RNE convert
    return __builtin_bit_cast(unsigned short, h);
}
__device__ __forceinline__ float sigmoidf(float x) { return 1.0f / (1.0f + __expf(-x)); }

// wave32 all-lane sum
__device__ __forceinline__ float waveRed(float v) {
#pragma unroll
    for (int off = 16; off > 0; off >>= 1) v += __shfl_xor(v, off, 32);
    return v;
}

extern "C" __global__ void __launch_bounds__(256)
ema_fused_kernel(const float* __restrict__ x,  const float* __restrict__ w1,
                 const float* __restrict__ b1, const float* __restrict__ w3,
                 const float* __restrict__ b3, const float* __restrict__ gnw,
                 const float* __restrict__ gnb, float* __restrict__ out)
{
    // ---- LDS: whole group resident (CDNA5: 320KB/WGP) ----
    __shared__ __align__(16) float s_gx[66 * 66 * CG];        // fp32 tile + zero halo (139392 B)
    __shared__ unsigned short s_x1[HWN * CG];                  // bf16 GN output     (65536 B)
    __shared__ unsigned short s_x2[HWN * CG];                  // bf16 conv output   (65536 B)
    __shared__ float s_xh[CG * Hh], s_xw[CG * Ww];             // row/col means
    __shared__ float s_sh[CG * Hh], s_sw[CG * Ww];             // sigmoid gates
    __shared__ float s_w1[CG * CG], s_b1[CG], s_w3[CG * CG * 9], s_b3[CG];
    __shared__ float s_gnw[CG], s_gnb[CG];
    __shared__ float s_sum[CG], s_sqs[CG], s_x1s[CG], s_x2s[CG];
    __shared__ float s_a[CG], s_bb[CG], s_sm1[CG], s_sm2[CG];

    const int tid  = threadIdx.x;
    const int n    = blockIdx.x;          // group index, 0..1023
    const int bb_  = n >> 5;              // batch
    const int gg   = n & 31;              // group within batch
    const int lane = tid & 31;
    const int wv   = tid >> 5;            // wave id, 0..7
    const int hh   = lane >> 4;           // lane half (WMMA layout)

    // ---- stage params, zero halo + accumulators ----
    for (int i = tid; i < CG * CG;     i += 256) s_w1[i] = w1[i];
    for (int i = tid; i < CG * CG * 9; i += 256) s_w3[i] = w3[i];
    if (tid < CG) {
        s_b1[tid] = b1[tid]; s_b3[tid] = b3[tid];
        s_gnw[tid] = gnw[tid]; s_gnb[tid] = gnb[tid];
        s_sum[tid] = 0.f; s_sqs[tid] = 0.f; s_x1s[tid] = 0.f; s_x2s[tid] = 0.f;
    }
    for (int i = tid; i < 66 * 66 * CG; i += 256) s_gx[i] = 0.f;
    __syncthreads();

    // ---- async copy gx tile: memory -> LDS directly (ASYNCcnt-tracked) ----
    // 8 group channels are 32 contiguous bytes per pixel -> 2x async b128.
    const float* xb = x + ((size_t)bb_ * HWN) * CCH + gg * CG;
#pragma unroll
    for (int i = 0; i < 16; ++i) {
        int p = tid + (i << 8);
        int y = p >> 6, xx = p & 63;
        unsigned lds0 = (unsigned)(size_t)(const void*)&s_gx[((y + 1) * 66 + (xx + 1)) * CG];
        unsigned long long ga = (unsigned long long)(size_t)(const void*)(xb + (size_t)p * CCH);
        // IOFFSET is added to both the LDS and global addresses (ISA 08 §4.4)
        asm volatile("global_load_async_to_lds_b128 %0, %1, off\n\t"
                     "global_load_async_to_lds_b128 %0, %1, off offset:16"
                     :: "v"(lds0), "v"(ga) : "memory");
    }
    asm volatile("s_wait_asynccnt 0" ::: "memory");
    __syncthreads();

    // ---- x_h[c][y], x_w[c][x] ----
    for (int r = tid; r < CG * Hh; r += 256) {
        int c = r >> 6, y = r & 63;
        float acc = 0.f;
        for (int xx = 0; xx < Ww; ++xx) acc += s_gx[((y + 1) * 66 + (xx + 1)) * CG + c];
        s_xh[c * Hh + y] = acc * (1.0f / 64.0f);
    }
    for (int r = tid; r < CG * Ww; r += 256) {
        int c = r >> 6, xx = r & 63;
        float acc = 0.f;
        for (int y = 0; y < Hh; ++y) acc += s_gx[((y + 1) * 66 + (xx + 1)) * CG + c];
        s_xw[c * Ww + xx] = acc * (1.0f / 64.0f);
    }
    __syncthreads();

    // ---- hwm = w1 @ cat + b1 ; sigmoid gates ----
    for (int e = tid; e < CG * 128; e += 256) {
        int o = e >> 7, q = e & 127;
        float acc = s_b1[o];
#pragma unroll
        for (int i = 0; i < CG; ++i) {
            float cv = (q < Hh) ? s_xh[i * Hh + q] : s_xw[i * Ww + (q - Hh)];
            acc = fmaf(s_w1[o * CG + i], cv, acc);
        }
        float sg = sigmoidf(acc);
        if (q < Hh) s_sh[o * Hh + q] = sg; else s_sw[o * Ww + (q - Hh)] = sg;
    }
    __syncthreads();

    // ---- gated stats (per-channel mean/var over plane) ----
    float psum[CG], psq[CG];
#pragma unroll
    for (int c = 0; c < CG; ++c) { psum[c] = 0.f; psq[c] = 0.f; }
#pragma unroll
    for (int i = 0; i < 16; ++i) {
        int p = tid + (i << 8);
        int y = p >> 6, xx = p & 63;
        const float* gp = &s_gx[((y + 1) * 66 + (xx + 1)) * CG];
#pragma unroll
        for (int c = 0; c < CG; ++c) {
            float gv = gp[c] * s_sh[c * Hh + y] * s_sw[c * Ww + xx];
            psum[c] += gv; psq[c] += gv * gv;
        }
    }
#pragma unroll
    for (int c = 0; c < CG; ++c) {
        float s = waveRed(psum[c]);
        float q = waveRed(psq[c]);
        if (lane == 0) { atomicAdd(&s_sum[c], s); atomicAdd(&s_sqs[c], q); }
    }
    __syncthreads();
    if (tid < CG) {
        float mu  = s_sum[tid] * (1.0f / 4096.0f);
        float var = s_sqs[tid] * (1.0f / 4096.0f) - mu * mu;
        float a   = s_gnw[tid] * rsqrtf(var + EPSV);
        s_a[tid] = a; s_bb[tid] = s_gnb[tid] - mu * a;
    }
    __syncthreads();

    // ---- x1 = GN(gated) -> bf16 LDS; accumulate channel sums ----
#pragma unroll
    for (int c = 0; c < CG; ++c) psum[c] = 0.f;
#pragma unroll
    for (int i = 0; i < 16; ++i) {
        int p = tid + (i << 8);
        int y = p >> 6, xx = p & 63;
        const float* gp = &s_gx[((y + 1) * 66 + (xx + 1)) * CG];
        unsigned short* xp = &s_x1[p * CG];
#pragma unroll
        for (int c = 0; c < CG; ++c) {
            float gv  = gp[c] * s_sh[c * Hh + y] * s_sw[c * Ww + xx];
            float x1v = fmaf(gv, s_a[c], s_bb[c]);
            xp[c] = f2bf(x1v);
            psum[c] += x1v;
        }
    }
#pragma unroll
    for (int c = 0; c < CG; ++c) {
        float s = waveRed(psum[c]);
        if (lane == 0) atomicAdd(&s_x1s[c], s);
    }

    // ---- 3x3 conv as implicit GEMM via v_wmma_f32_16x16x32_bf16 ----
    // K = tap*8 + ci (tap = kh*3+kw; 72 real, padded to 96 -> 3 WMMAs of K=32).
    // 16-bit A 16x32 layout: lane-half hh=0 owns K [0..7]+[16..23] of each block,
    // hh=1 owns [8..15]+[24..31] -> per block, each lane gathers 2 taps x 8 channels
    // (branchless: block 2 uses an always-valid tap-8 load times a 0/1 lane mask).
    const int nn = lane & 15;
    const float nmask = (nn < CG) ? 1.f : 0.f;
    v16bf bfrag[3];
#pragma unroll
    for (int kb = 0; kb < 3; ++kb) {
#pragma unroll
        for (int e = 0; e < 16; ++e) {
            int K    = kb * 32 + hh * 8 + (e & 7) + ((e >> 3) << 4);
            int tap  = K >> 3;
            int ci   = K & 7;
            int tapc = (tap > 8) ? 8 : tap;                       // clamp -> always in-bounds
            float v  = s_w3[((nn & 7) * CG + ci) * 9 + tapc];
            v = (K < 72) ? (v * nmask) : 0.f;                     // cndmask, no exec branch
            bfrag[kb][e] = (__bf16)v;
        }
    }

    const int m = lane & 15;
    const float bias = s_b3[nn & 7];        // hoisted: loop-invariant, clamped index
    float x2part = 0.f;
    for (int it = 0; it < 32; ++it) {
        int T  = wv + (it << 3);            // tile id 0..255, 16 px per tile (quarter row)
        int y  = T >> 2;
        int px = ((T & 3) << 4) + m;        // this lane's A-row pixel (x coord, unpadded)
        // top-left halo corner of this pixel's 3x3 window
        const float* tile = &s_gx[(y * 66 + px) * CG];
        v8f acc = {0.f, 0.f, 0.f, 0.f, 0.f, 0.f, 0.f, 0.f};

#pragma unroll
        for (int kb = 0; kb < 2; ++kb) {
            int tapA = kb * 4 + hh;          // elements 0..7
            int tapB = kb * 4 + 2 + hh;      // elements 8..15
            int tyA = tapA / 3, txA = tapA % 3;
            int tyB = tapB / 3, txB = tapB % 3;
            const float4* pA = (const float4*)(tile + (tyA * 66 + txA) * CG);
            const float4* pB = (const float4*)(tile + (tyB * 66 + txB) * CG);
            float4 a0 = pA[0], a1 = pA[1];
            float4 b0 = pB[0], b1 = pB[1];
            v16bf af;
            af[0]  = (__bf16)a0.x; af[1]  = (__bf16)a0.y; af[2]  = (__bf16)a0.z; af[3]  = (__bf16)a0.w;
            af[4]  = (__bf16)a1.x; af[5]  = (__bf16)a1.y; af[6]  = (__bf16)a1.z; af[7]  = (__bf16)a1.w;
            af[8]  = (__bf16)b0.x; af[9]  = (__bf16)b0.y; af[10] = (__bf16)b0.z; af[11] = (__bf16)b0.w;
            af[12] = (__bf16)b1.x; af[13] = (__bf16)b1.y; af[14] = (__bf16)b1.z; af[15] = (__bf16)b1.w;
            acc = __builtin_amdgcn_wmma_f32_16x16x32_bf16(
                      false, af, false, bfrag[kb], (short)0, acc, false, false);
        }
        {   // K-block 2: only tap 8 (kh=2,kw=2), hh==0 lanes; rest zero
            const float4* pA = (const float4*)(tile + (2 * 66 + 2) * CG);
            float4 a0 = pA[0], a1 = pA[1];
            float msk = hh ? 0.f : 1.f;
            v16bf af;
            af[0] = (__bf16)(a0.x * msk); af[1] = (__bf16)(a0.y * msk);
            af[2] = (__bf16)(a0.z * msk); af[3] = (__bf16)(a0.w * msk);
            af[4] = (__bf16)(a1.x * msk); af[5] = (__bf16)(a1.y * msk);
            af[6] = (__bf16)(a1.z * msk); af[7] = (__bf16)(a1.w * msk);
#pragma unroll
            for (int e = 8; e < 16; ++e) af[e] = (__bf16)0.f;
            acc = __builtin_amdgcn_wmma_f32_16x16x32_bf16(
                      false, af, false, bfrag[2], (short)0, acc, false, false);
        }

        // D layout: VGPR r -> M = r + 8*hh, N = lane&15
        if (nn < CG) {
            unsigned short* xo = &s_x2[(((T << 4) + 8 * hh)) * CG + nn];
#pragma unroll
            for (int r = 0; r < 8; ++r) {
                float v = acc[r] + bias;
                xo[r * CG] = f2bf(v);
                x2part += v;
            }
        }
    }
    if (nn < CG) atomicAdd(&s_x2s[nn], x2part);
    __syncthreads();

    // ---- channel softmaxes of plane means ----
    if (tid == 0) {
        float v1[CG], v2[CG];
        float m1 = -1e30f, m2 = -1e30f;
        for (int c = 0; c < CG; ++c) {
            v1[c] = s_x1s[c] * (1.0f / 4096.0f);
            v2[c] = s_x2s[c] * (1.0f / 4096.0f);
            m1 = fmaxf(m1, v1[c]); m2 = fmaxf(m2, v2[c]);
        }
        float d1 = 0.f, d2 = 0.f;
        for (int c = 0; c < CG; ++c) {
            v1[c] = __expf(v1[c] - m1); d1 += v1[c];
            v2[c] = __expf(v2[c] - m2); d2 += v2[c];
        }
        float r1 = 1.0f / d1, r2 = 1.0f / d2;
        for (int c = 0; c < CG; ++c) { s_sm1[c] = v1[c] * r1; s_sm2[c] = v2[c] * r2; }
    }
    __syncthreads();

    // ---- weights = sm(x1mean) . x2 + sm(x2mean) . x1 ; out = gx * sigmoid(weights) ----
    float* ob = out + ((size_t)bb_ * HWN) * CCH + gg * CG;
#pragma unroll
    for (int i = 0; i < 16; ++i) {
        int p = tid + (i << 8);
        int y = p >> 6, xx = p & 63;
        const unsigned short* x2p = &s_x2[p * CG];
        const unsigned short* x1p = &s_x1[p * CG];
        float wsum = 0.f;
#pragma unroll
        for (int c = 0; c < CG; ++c)
            wsum += s_sm1[c] * bf2f(x2p[c]) + s_sm2[c] * bf2f(x1p[c]);
        float sg = sigmoidf(wsum);
        const float* gp = &s_gx[((y + 1) * 66 + (xx + 1)) * CG];
        float4 o0, o1;
        o0.x = gp[0] * sg; o0.y = gp[1] * sg; o0.z = gp[2] * sg; o0.w = gp[3] * sg;
        o1.x = gp[4] * sg; o1.y = gp[5] * sg; o1.z = gp[6] * sg; o1.w = gp[7] * sg;
        float4* dst = (float4*)(ob + (size_t)p * CCH);
        dst[0] = o0; dst[1] = o1;
    }
}

extern "C" void kernel_launch(void* const* d_in, const int* in_sizes, int n_in,
                              void* d_out, int out_size, void* d_ws, size_t ws_size,
                              hipStream_t stream) {
    const float* x   = (const float*)d_in[0];
    const float* w1  = (const float*)d_in[1];
    const float* b1  = (const float*)d_in[2];
    const float* w3  = (const float*)d_in[3];
    const float* b3  = (const float*)d_in[4];
    const float* gnw = (const float*)d_in[5];
    const float* gnb = (const float*)d_in[6];
    float* out = (float*)d_out;
    (void)in_sizes; (void)n_in; (void)out_size; (void)d_ws; (void)ws_size;
    // one workgroup per (batch, group) = 32*32 = 1024 groups
    ema_fused_kernel<<<dim3(1024), dim3(256), 0, stream>>>(x, w1, b1, w3, b3, gnw, gnb, out);
}
